// MultiHeadAttention_30279519436842
// MI455X (gfx1250) — compile-verified, matrix operand order checked by reading, not run on hardware
//
#include <hip/hip_runtime.h>
#include <hip/hip_bf16.h>

// ---------------------------------------------------------------------------
// MHA forward for MI455X (gfx1250, wave32, WMMA).
//   B=4, L=2048, D=1024, H=16, HD=64.
// 5 launches: Q/K/V projection GEMMs (bf16 WMMA, fp32 accum), flash attention
// (bf16 WMMA QK^T and PV with fp32 online softmax), output projection GEMM.
// ---------------------------------------------------------------------------

typedef __attribute__((ext_vector_type(16))) __bf16 v16bf;
typedef __attribute__((ext_vector_type(8)))  float  v8f;
typedef int v4i __attribute__((vector_size(16)));   // matches builtin param type
typedef unsigned short bfraw; // host-safe alias for bf16 storage

constexpr int Bc = 4, Lc = 2048, Dc = 1024, Hc = 16, HDc = 64;

#define DEV static __device__ __forceinline__

// CDNA5 async direct-to-LDS loads (ASYNCcnt path), if the toolchain has them.
// Probe round confirmed: arity 4, param0 = v4i addrspace(1)* (global src).
#if __has_builtin(__builtin_amdgcn_global_load_async_to_lds_b128) && \
    __has_builtin(__builtin_amdgcn_s_wait_asynccnt)
#define USE_ASYNC_LDS 1
#define AS_GLOBAL(p) ((__attribute__((address_space(1))) v4i*)(void*)(p))
#define AS_LOCAL(p)  ((__attribute__((address_space(3))) v4i*)(void*)(p))
#endif

// ---------------------------------------------------------------------------
// WMMA helpers
// ---------------------------------------------------------------------------
DEV v8f wmma_bf16(v16bf a, v16bf b, v8f c) {
  // V_WMMA_F32_16X16X32_BF16: D = A(16x32) * B(32x16) + C(16x16 f32)
  return __builtin_amdgcn_wmma_f32_16x16x32_bf16(
      /*neg_a=*/false, a, /*neg_b=*/false, b,
      /*c_mod=*/(short)0, c, /*reuse_a=*/false, /*reuse_b=*/false);
}

// Load a 16x32 bf16 A/B fragment from a row-major [rows][strideElems] array
// (LDS or global). Per ISA 7.12.2 (16-bit A 16x32): lanes 0-15 hold row
// (rowBase+lane), K = {0..7, 16..23}; lanes 16-31 same rows, K shifted +8.
// Each lane: two contiguous 16-byte chunks.
DEV v16bf load_frag(const __bf16* base, int strideElems, int rowBase, int kOff) {
  const int lane = threadIdx.x & 31;
  const int lo   = lane & 15;
  const int hi8  = (lane >> 4) << 3;          // 0 or 8
  const __bf16* p = base + (size_t)(rowBase + lo) * strideElems + kOff + hi8;
  union { uint4 u[2]; v16bf v; } t;
  t.u[0] = *(const uint4*)(p);                // halves 0..7  -> K = kOff+hi8+0..7
  t.u[1] = *(const uint4*)(p + 16);           // halves 8..15 -> K = kOff+hi8+16..23
  return t.v;
}

DEV uint2 pack4bf(float4 f) {
  union { __bf16 h[4]; uint2 q; } t;
  t.h[0] = (__bf16)f.x; t.h[1] = (__bf16)f.y;
  t.h[2] = (__bf16)f.z; t.h[3] = (__bf16)f.w;
  return t.q;
}

// ---------------------------------------------------------------------------
// GEMM: out = A[M,K](f32) * Bw[K,N](f32) + bias[N], computed in bf16 WMMA.
// MODE 0: store f32 [M,N]
// MODE 1: store bf16 head-split  [B,H,L,HD]   (for Q, K)
// MODE 2: store bf16 head-transposed [B,H,HD,L] (for V; PV wants B^T layout)
// Block: 256 threads = 8 waves; tile 128(M) x 64(N), K-step 64.
// ---------------------------------------------------------------------------
constexpr int BM = 128, BN = 64, BK = 64;
constexpr int ASTR = 72, BSTR = 72; // padded LDS strides (bank-conflict free)

template <int MODE>
__global__ __launch_bounds__(256) void gemm_bf16_wmma(
    const float* __restrict__ A, const float* __restrict__ Bw,
    const float* __restrict__ bias, void* __restrict__ outp,
    int M, int N, int K) {
  __shared__ __align__(16) __bf16 As[BM * ASTR];
  __shared__ __align__(16) __bf16 Bs[BN * BSTR];

  const int tid  = threadIdx.x;
  const int wid  = tid >> 5;
  const int lane = tid & 31;
  const int lo   = lane & 15;
  const int hi8  = (lane >> 4) << 3;
  const int mTile = blockIdx.y * BM;
  const int nTile = blockIdx.x * BN;

  v8f acc[4] = {};

  for (int k0 = 0; k0 < K; k0 += BK) {
    if (k0 + BK < K) { // prefetch next A tile into cache (global_prefetch_b8)
      const int prow = tid >> 1, pcol = (tid & 1) * 32;
      __builtin_prefetch(&A[(size_t)(mTile + prow) * K + k0 + BK + pcol], 0, 0);
    }
    // stage A tile: 128x64 f32 -> bf16 LDS
#pragma unroll
    for (int it = 0; it < 8; ++it) {
      const int u = tid + it * 256;
      const int row = u >> 4, c4 = (u & 15) << 2;
      float4 f = *(const float4*)&A[(size_t)(mTile + row) * K + k0 + c4];
      *(uint2*)&As[row * ASTR + c4] = pack4bf(f);
    }
    // stage B tile transposed: f32 [K,N] -> bf16 LDS [n][k]
#pragma unroll
    for (int it = 0; it < 4; ++it) {
      const int u = tid + it * 256;
      const int kk = u >> 4, c4 = (u & 15) << 2;
      float4 f = *(const float4*)&Bw[(size_t)(k0 + kk) * N + nTile + c4];
      Bs[(c4 + 0) * BSTR + kk] = (__bf16)f.x;
      Bs[(c4 + 1) * BSTR + kk] = (__bf16)f.y;
      Bs[(c4 + 2) * BSTR + kk] = (__bf16)f.z;
      Bs[(c4 + 3) * BSTR + kk] = (__bf16)f.w;
    }
    __syncthreads();
#pragma unroll
    for (int ks = 0; ks < BK; ks += 32) {
      v16bf aF = load_frag(As, ASTR, wid * 16, ks);
#pragma unroll
      for (int j = 0; j < 4; ++j) {
        v16bf bF = load_frag(Bs, BSTR, j * 16, ks);
        acc[j] = wmma_bf16(aF, bF, acc[j]);
      }
    }
    __syncthreads();
  }

  // Epilogue. C layout (ISA 7.12.2): VGPR r -> row r + 8*(lane>=16), col lane%16.
#pragma unroll
  for (int j = 0; j < 4; ++j) {
    const int n = nTile + j * 16 + lo;
    const float bv = bias[n];
#pragma unroll
    for (int r = 0; r < 8; ++r) {
      const int m = mTile + wid * 16 + r + hi8;
      const float val = acc[j][r] + bv;
      if (MODE == 0) {
        ((float*)outp)[(size_t)m * N + n] = val;
      } else {
        const int b = m / Lc, l = m % Lc;
        const int h = n / HDc, d = n % HDc;
        __bf16* ob = (__bf16*)outp;
        if (MODE == 1)
          ob[(((size_t)(b * Hc + h)) * Lc + l) * HDc + d] = (__bf16)val;
        else
          ob[(((size_t)(b * Hc + h)) * HDc + d) * Lc + l] = (__bf16)val;
      }
    }
  }
}

// ---------------------------------------------------------------------------
// Flash attention: per block (4 waves) one (b,h) and 64 q rows; stream over
// key tiles of 64 with online softmax. All matmuls via bf16 WMMA.
// Qh/Kh: bf16 [B,H,L,64]; Vt: bf16 [B,H,64,L]; mask: f32 [B,1,L,L];
// attn out: f32 [B,L,D] (merged heads) for the final projection GEMM.
// K/V tiles use async direct-to-LDS (ASYNCcnt) when available; the 64x64
// mask tile is staged through LDS with coalesced b128 loads (mask reuse is
// served from the 192MB L2 across the 16 heads).
// ---------------------------------------------------------------------------
constexpr int KT = 64, QT = 64;
constexpr int KSTR = 72, VSTR = 72, PSTR = 72, MSTR = 68;

__global__ __launch_bounds__(128) void flash_attn_wmma(
    const bfraw* __restrict__ Qh_, const bfraw* __restrict__ Kh_,
    const bfraw* __restrict__ Vt_, const float* __restrict__ mask,
    float* __restrict__ attn) {
  __shared__ __align__(16) __bf16 Ks[KT * KSTR];
  __shared__ __align__(16) __bf16 Vs[HDc * VSTR];
  __shared__ __align__(16) __bf16 Ps[4 * 16 * PSTR]; // wave-private P tiles
  __shared__ __align__(16) float  Ms[QT * MSTR];     // mask tile (f32)

  const int tid  = threadIdx.x;
  const int wid  = tid >> 5;
  const int lane = tid & 31;
  const int lo   = lane & 15;
  const int hi8  = (lane >> 4) << 3;
  const int bh = blockIdx.y;
  const int b = bh / Hc, h = bh % Hc;
  const int qBlk0 = blockIdx.x * QT;
  const int qRow0 = qBlk0 + wid * 16;

  const __bf16* Qb = (const __bf16*)Qh_ + (size_t)bh * Lc * HDc;
  const __bf16* Kb = (const __bf16*)Kh_ + (size_t)bh * Lc * HDc;
  const __bf16* Vb = (const __bf16*)Vt_ + (size_t)bh * HDc * Lc;
  const float*  Mb = mask + (size_t)b * Lc * Lc;

  // Q fragments live in registers for the whole pass (HD=64 -> 2 K-chunks)
  const v16bf qf0 = load_frag(Qb, HDc, qRow0, 0);
  const v16bf qf1 = load_frag(Qb, HDc, qRow0, 32);

  v8f o[4] = {};
  float rowMax[8], rowSum[8];
#pragma unroll
  for (int r = 0; r < 8; ++r) { rowMax[r] = -3.0e38f; rowSum[r] = 0.f; }

  const float scale = 0.125f;   // 1/sqrt(64)
  const float NEGV  = -1.0e15f;

  for (int kt = 0; kt < Lc / KT; ++kt) {
    const int kBase = kt * KT;
    __syncthreads();
#if defined(USE_ASYNC_LDS)
    // stage K/V tiles via GLOBAL_LOAD_ASYNC_TO_LDS_B128 (no VGPR round-trip)
#pragma unroll
    for (int it = 0; it < 4; ++it) {
      const int u = tid + it * 128;
      const int row = u >> 3, c8 = (u & 7) << 3;
      __builtin_amdgcn_global_load_async_to_lds_b128(
          AS_GLOBAL(&Kb[(size_t)(kBase + row) * HDc + c8]),
          AS_LOCAL(&Ks[row * KSTR + c8]), 0, 0);
      __builtin_amdgcn_global_load_async_to_lds_b128(
          AS_GLOBAL(&Vb[(size_t)row * Lc + kBase + c8]),
          AS_LOCAL(&Vs[row * VSTR + c8]), 0, 0);
    }
#else
    // fallback: sync staging through VGPRs
#pragma unroll
    for (int it = 0; it < 4; ++it) {
      const int u = tid + it * 128;
      const int row = u >> 3, c8 = (u & 7) << 3;
      *(uint4*)&Ks[row * KSTR + c8] =
          *(const uint4*)&Kb[(size_t)(kBase + row) * HDc + c8];
      *(uint4*)&Vs[row * VSTR + c8] =
          *(const uint4*)&Vb[(size_t)row * Lc + kBase + c8];
    }
#endif
    // stage mask tile: 64x64 f32, coalesced b128 (replaces 32 scattered
    // dword loads per lane with 8 vector loads per thread)
#pragma unroll
    for (int it = 0; it < 8; ++it) {
      const int u = tid + it * 128;
      const int row = u >> 4, c4 = (u & 15) << 2;
      *(float4*)&Ms[row * MSTR + c4] =
          *(const float4*)&Mb[(size_t)(qBlk0 + row) * Lc + kBase + c4];
    }
#if defined(USE_ASYNC_LDS)
    __builtin_amdgcn_s_wait_asynccnt(0);
#endif
    __syncthreads();

    // S = Q * K^T : 16 q-rows x 64 keys
    v8f s[4] = {};
#pragma unroll
    for (int j = 0; j < 4; ++j) {
      s[j] = wmma_bf16(qf0, load_frag(Ks, KSTR, j * 16, 0),  s[j]);
      s[j] = wmma_bf16(qf1, load_frag(Ks, KSTR, j * 16, 32), s[j]);
    }

    // scale + additive mask (from LDS)
    float vals[4][8];
#pragma unroll
    for (int r = 0; r < 8; ++r) {
      const float* mrow = &Ms[(wid * 16 + r + hi8) * MSTR + lo];
#pragma unroll
      for (int j = 0; j < 4; ++j)
        vals[j][r] = s[j][r] * scale + mrow[j * 16] * NEGV;
    }

    // online softmax (rows live across the 16 lanes of each wave half)
#pragma unroll
    for (int r = 0; r < 8; ++r) {
      float lmax = fmaxf(fmaxf(vals[0][r], vals[1][r]),
                         fmaxf(vals[2][r], vals[3][r]));
#pragma unroll
      for (int m = 1; m < 16; m <<= 1) lmax = fmaxf(lmax, __shfl_xor(lmax, m, 32));
      const float nmax = fmaxf(rowMax[r], lmax);
      const float corr = __expf(rowMax[r] - nmax);
      rowMax[r] = nmax;
      float lsum = 0.f;
#pragma unroll
      for (int j = 0; j < 4; ++j) {
        const float p = __expf(vals[j][r] - nmax);
        lsum += p;
        // re-layout C->A fragment via wave-private LDS
        Ps[(wid * 16 + r + hi8) * PSTR + j * 16 + lo] = (__bf16)p;
      }
#pragma unroll
      for (int m = 1; m < 16; m <<= 1) lsum += __shfl_xor(lsum, m, 32);
      rowSum[r] = rowSum[r] * corr + lsum;
#pragma unroll
      for (int jd = 0; jd < 4; ++jd) o[jd][r] *= corr;
    }

    // O += P * V  (K dim = 64 keys -> 2 chunks; N dim = 64 head dims)
#pragma unroll
    for (int ks = 0; ks < KT; ks += 32) {
      v16bf pf = load_frag(Ps, PSTR, wid * 16, ks);
#pragma unroll
      for (int jd = 0; jd < 4; ++jd)
        o[jd] = wmma_bf16(pf, load_frag(Vs, VSTR, jd * 16, ks), o[jd]);
    }
  }

  // normalize and write merged-head f32 [B,L,D]
#pragma unroll
  for (int jd = 0; jd < 4; ++jd) {
#pragma unroll
    for (int r = 0; r < 8; ++r) {
      const int l = qRow0 + r + hi8;
      const int dcol = h * HDc + jd * 16 + lo;
      attn[((size_t)b * Lc + l) * Dc + dcol] = o[jd][r] / rowSum[r];
    }
  }
}

// ---------------------------------------------------------------------------
extern "C" void kernel_launch(void* const* d_in, const int* in_sizes, int n_in,
                              void* d_out, int out_size, void* d_ws, size_t ws_size,
                              hipStream_t stream) {
  const float* q    = (const float*)d_in[0];
  const float* k    = (const float*)d_in[1];
  const float* v    = (const float*)d_in[2];
  const float* mask = (const float*)d_in[3];
  const float* Wq   = (const float*)d_in[4];
  const float* bq   = (const float*)d_in[5];
  const float* Wk   = (const float*)d_in[6];
  const float* bk   = (const float*)d_in[7];
  const float* Wv   = (const float*)d_in[8];
  const float* bv   = (const float*)d_in[9];
  const float* Wo   = (const float*)d_in[10];
  const float* bo   = (const float*)d_in[11];

  const size_t E = (size_t)Bc * Lc * Dc; // 8.4M elements
  char* ws = (char*)d_ws;
  bfraw* Qh  = (bfraw*)(ws);                 // 16 MB bf16 [B,H,L,64]
  bfraw* Kh  = (bfraw*)(ws + 2 * E);         // 16 MB bf16 [B,H,L,64]
  bfraw* Vt  = (bfraw*)(ws + 4 * E);         // 16 MB bf16 [B,H,64,L]
  float* attn = (float*)(ws + 6 * E);        // 32 MB f32  [B,L,D]

  const int M = Bc * Lc, N = Dc, K = Dc;
  dim3 gGrid(N / BN, M / BM), gBlk(256);

  gemm_bf16_wmma<1><<<gGrid, gBlk, 0, stream>>>(q, Wq, bq, (void*)Qh, M, N, K);
  gemm_bf16_wmma<1><<<gGrid, gBlk, 0, stream>>>(k, Wk, bk, (void*)Kh, M, N, K);
  gemm_bf16_wmma<2><<<gGrid, gBlk, 0, stream>>>(v, Wv, bv, (void*)Vt, M, N, K);

  flash_attn_wmma<<<dim3(Lc / QT, Bc * Hc), dim3(128), 0, stream>>>(
      Qh, Kh, Vt, mask, attn);

  gemm_bf16_wmma<0><<<gGrid, gBlk, 0, stream>>>(attn, Wo, bo, d_out, M, N, K);
}
